// LocationAwareAttention_32727650795941
// MI455X (gfx1250) — matrix operand order repaired and checked by reading, not compiled, not run on hardware
//
#include <hip/hip_runtime.h>
#include <hip/hip_bf16.h>
#include <stdint.h>

// ---------------- problem constants ----------------
#define B    32
#define T    4096
#define DEC  1024
#define ENC  1024
#define ATTN 512
#define KCH  10

// ---------------- tiling ----------------
#define TM    64            // t-rows per workgroup
#define ECH   128           // e (K-dim) super-chunk staged per async round
#define NCH   (ENC / ECH)   // 8 super-chunks
#define ROWP  (ENC + 8)     // bf16 LDS row pitch (pad 16B to break bank conflicts)

#define SVAL_BYTES  (TM * ROWP * 2)          // 132096
#define STAGE_BYTES (2 * TM * ECH * 4)       // 65536 (double buffered f32 stage)
#define SMEM_MAIN   (SVAL_BYTES + STAGE_BYTES + (68 + 64 + 64) * 4)  // 198416

// ---------------- types ----------------
typedef __bf16   v16bf __attribute__((ext_vector_type(16)));
typedef float    v8f   __attribute__((ext_vector_type(8)));
typedef int      v4i   __attribute__((ext_vector_type(4)));
typedef unsigned uint4v __attribute__((ext_vector_type(4)));
typedef unsigned uint2v __attribute__((ext_vector_type(2)));
typedef float    float4v __attribute__((ext_vector_type(4)));

struct U32x8 { uint4v lo, hi; };

__device__ __forceinline__ v16bf make_frag(uint4v lo, uint4v hi) {
  U32x8 t; t.lo = lo; t.hi = hi;
  return __builtin_bit_cast(v16bf, t);
}

__device__ __forceinline__ unsigned f2bf1(float f) {            // RNE f32 -> bf16 bits
  unsigned u = __float_as_uint(f);
  return (u + 0x7fffu + ((u >> 16) & 1u)) >> 16;
}
__device__ __forceinline__ unsigned pk2(float a, float b) { return f2bf1(a) | (f2bf1(b) << 16); }
__device__ __forceinline__ float bf2f(unsigned bits) { return __uint_as_float(bits << 16); }

// ---------------- CDNA5 async-to-LDS helpers ----------------
__device__ __forceinline__ void async_load16(const void* g, void* lds_generic) {
#if __has_builtin(__builtin_amdgcn_global_load_async_to_lds_b128)
  // signature (probe-discovered): void(v4i AS1*, v4i AS3*, imm offset, imm cpol)
  __builtin_amdgcn_global_load_async_to_lds_b128(
      (__attribute__((address_space(1))) v4i*)g,
      (__attribute__((address_space(3))) v4i*)lds_generic, 0, 0);
#else
  unsigned loff = (unsigned)(unsigned long long)
      (__attribute__((address_space(3))) char*)lds_generic;
  asm volatile("global_load_async_to_lds_b128 %0, %1, off"
               :: "v"(loff), "v"((unsigned long long)(uintptr_t)g)
               : "memory");
#endif
}

__device__ __forceinline__ void wait_async0() {
#if __has_builtin(__builtin_amdgcn_s_wait_asynccnt)
  __builtin_amdgcn_s_wait_asynccnt(0);
#else
  asm volatile("s_wait_asynccnt 0" ::: "memory");
#endif
}

// =====================================================================
// Prep 1: WV (f32 [ATTN][ENC]) -> bf16, same row-major layout
// =====================================================================
__global__ void prep_wv_bf16(const float* __restrict__ WV,
                             unsigned short* __restrict__ WVb, int n4) {
  int i = blockIdx.x * blockDim.x + threadIdx.x;
  if (i >= n4) return;
  float4v f = ((const float4v*)WV)[i];
  uint2v o; o.x = pk2(f.x, f.y); o.y = pk2(f.z, f.w);
  ((uint2v*)WVb)[i] = o;
}

// =====================================================================
// Prep 2: qb2[b][a] = query[b]·WQ[a] + bias[a] + Σk WU[a,k]·conv_b[k]
//         c_d[a]    = Σk WU[a,k]·conv_w[k,0,d]      (d = 0,1,2)
// =====================================================================
__global__ void prep_qbc(const float* __restrict__ query, const float* __restrict__ WQ,
                         const float* __restrict__ bias,  const float* __restrict__ WU,
                         const float* __restrict__ conv_w, const float* __restrict__ conv_b,
                         float* __restrict__ qb2, float* __restrict__ c0,
                         float* __restrict__ c1,  float* __restrict__ c2) {
  int idx = blockIdx.x * blockDim.x + threadIdx.x;   // B*ATTN threads
  if (idx >= B * ATTN) return;
  int b = idx / ATTN, a = idx % ATTN;

  const float4v* q4 = (const float4v*)(query + (size_t)b * DEC);
  const float4v* w4 = (const float4v*)(WQ + (size_t)a * DEC);
  float s = 0.f;
  for (int i = 0; i < DEC / 4; ++i) {
    float4v q = q4[i], w = w4[i];
    s += q.x * w.x + q.y * w.y + q.z * w.z + q.w * w.w;
  }
  float ucb = 0.f;
  for (int k = 0; k < KCH; ++k) ucb += WU[(size_t)a * KCH + k] * conv_b[k];
  qb2[idx] = s + bias[a] + ucb;

  if (b == 0) {
    float s0 = 0.f, s1 = 0.f, s2 = 0.f;
    for (int k = 0; k < KCH; ++k) {
      float wu = WU[(size_t)a * KCH + k];
      s0 += wu * conv_w[k * 3 + 0];
      s1 += wu * conv_w[k * 3 + 1];
      s2 += wu * conv_w[k * 3 + 2];
    }
    c0[a] = s0; c1[a] = s1; c2[a] = s2;
  }
}

// =====================================================================
// Prep 3: zero context accumulators + denominators
// =====================================================================
__global__ void prep_zero(float* __restrict__ ctx, float* __restrict__ denom) {
  int i = blockIdx.x * blockDim.x + threadIdx.x;
  if (i < B * ENC) ctx[i] = 0.f;
  if (i < B) denom[i] = 0.f;
}

// =====================================================================
// Main fused kernel: one workgroup = (b, 64-row t-tile)
//   value tile -> (async b128) LDS f32 stage -> bf16 LDS tile
//   v_wmma_f32_16x16x32_bf16 GEMM vs WVb, tanh/fc/sigmoid epilogue,
//   unnormalized context + denom accumulation via global f32 atomics.
// =====================================================================
__global__ void __launch_bounds__(256)
law_main(const float* __restrict__ value, const float* __restrict__ last_align,
         const float* __restrict__ fc_w,
         const unsigned short* __restrict__ WVb, const float* __restrict__ qb2,
         const float* __restrict__ c0, const float* __restrict__ c1,
         const float* __restrict__ c2,
         float* __restrict__ scores, float* __restrict__ ctx,
         float* __restrict__ denom) {
  extern __shared__ char smem[];
  unsigned short* sVal  = (unsigned short*)smem;                      // [TM][ROWP] bf16
  float*          sStage = (float*)(smem + SVAL_BYTES);               // [2][TM*ECH] f32
  float*          sLa    = (float*)(smem + SVAL_BYTES + STAGE_BYTES); // 66 (+2 pad)
  float*          sPart  = sLa + 68;                                  // 64
  float*          sScore = sPart + 64;                                // 64

  const int tid  = threadIdx.x;
  const int wave = tid >> 5;
  const int lane = tid & 31;
  const int b    = blockIdx.y;
  const int t0   = blockIdx.x * TM;

  if (tid < TM + 2) {
    int g = t0 - 1 + tid;
    sLa[tid] = (g >= 0 && g < T) ? last_align[(size_t)b * T + g] : 0.f;
  }
  if (tid < TM) sPart[tid] = 0.f;
  __syncthreads();

  const char* vbase = (const char*)value;

  // ---- issue async stage for super-chunk c into buffer buf ----
  auto issue_chunk = [&](int c, int buf) {
    size_t rowBytes = (size_t)ENC * 4;
    size_t gbase = ((size_t)(b * T + t0) * ENC + (size_t)c * ECH) * 4;
#pragma unroll
    for (int it = 0; it < 8; ++it) {
      int j = it * 256 + tid;          // 2048 x 16B chunks
      int row = j >> 5, inrow = j & 31; // 32 chunks per 128-f32 row
      const void* g = vbase + gbase + (size_t)row * rowBytes + (size_t)inrow * 16;
      void* l = smem + SVAL_BYTES + (size_t)buf * (TM * ECH * 4) + (size_t)j * 16;
      async_load16(g, l);
    }
  };

  // ---- convert staged f32 chunk -> bf16 tile columns [c*ECH, c*ECH+ECH) ----
  auto convert_chunk = [&](int c, int buf) {
    const float4v* st = (const float4v*)(sStage + buf * (TM * ECH));
#pragma unroll
    for (int it = 0; it < 8; ++it) {
      int j = it * 256 + tid;          // float4 index, 2048 total
      int row = j >> 5, k4 = j & 31;
      float4v f = st[j];
      uint2v o; o.x = pk2(f.x, f.y); o.y = pk2(f.z, f.w);
      *(uint2v*)(sVal + row * ROWP + c * ECH + k4 * 4) = o;
    }
  };

  v8f acc[4][4] = {};   // [m-subtile][a-tile]
  const int ln   = lane & 15;
  const int half = lane >> 4;

  issue_chunk(0, 0);

  for (int c = 0; c < NCH; ++c) {
    wait_async0();
    __syncthreads();
    if (c + 1 < NCH) issue_chunk(c + 1, (c + 1) & 1);
    convert_chunk(c, c & 1);
    __syncthreads();

#pragma unroll
    for (int kc = 0; kc < 4; ++kc) {
      const int kbase = c * ECH + kc * 32;
      // A fragments: 16-bit A 16x32 layout (lanes 0-15: K 0-7/16-23; 16-31: K 8-15/24-31)
      const int koffA = kbase + (half ? 8 : 0);
      v16bf af[4];
#pragma unroll
      for (int i = 0; i < 4; ++i) {
        const uint4v* p = (const uint4v*)(sVal + (i * 16 + ln) * ROWP + koffA);
        af[i] = make_frag(p[0], p[2]);
      }
      // B fragments: 16-bit B 32x16 (lanes 0-15: K 0-7/16-23; 16-31: K 8-15/24-31)
      const int koffB = kbase + (half ? 8 : 0);
#pragma unroll
      for (int j = 0; j < 4; ++j) {
        int a = wave * 64 + j * 16 + ln;
        const uint4v* p = (const uint4v*)(WVb + (size_t)a * ENC + koffB);
        v16bf bf = make_frag(p[0], p[2]);
#pragma unroll
        for (int i = 0; i < 4; ++i)
          acc[i][j] = __builtin_amdgcn_wmma_f32_16x16x32_bf16(
              false, af[i], false, bf, (short)0, acc[i][j], false, false);
      }
    }
  }
  __syncthreads();

  // ---- epilogue: score partials  sPart[t] += Σ_a tanh(v+q+u+bias)·fc_w[a] ----
  {
    float part[4][8];
#pragma unroll
    for (int i = 0; i < 4; ++i)
#pragma unroll
      for (int r = 0; r < 8; ++r) part[i][r] = 0.f;

#pragma unroll
    for (int j = 0; j < 4; ++j) {
      int a = wave * 64 + j * 16 + ln;
      float qb = qb2[(size_t)b * ATTN + a];
      float k0 = c0[a], k1 = c1[a], k2 = c2[a], fw = fc_w[a];
#pragma unroll
      for (int i = 0; i < 4; ++i) {
#pragma unroll
        for (int r = 0; r < 8; ++r) {
          int tl = i * 16 + half * 8 + r;
          float x = acc[i][j][r] + qb
                  + k0 * sLa[tl] + k1 * sLa[tl + 1] + k2 * sLa[tl + 2];
          part[i][r] += tanhf(x) * fw;
        }
      }
    }
#pragma unroll
    for (int i = 0; i < 4; ++i) {
#pragma unroll
      for (int r = 0; r < 8; ++r) {
        float v = part[i][r];
        v += __shfl_xor(v, 1, 32);
        v += __shfl_xor(v, 2, 32);
        v += __shfl_xor(v, 4, 32);
        v += __shfl_xor(v, 8, 32);
        if (ln == 0) atomicAdd(&sPart[i * 16 + half * 8 + r], v);
      }
    }
  }
  __syncthreads();

  // ---- sigmoid, emit raw scores ----
  if (tid < TM) {
    float s = 1.f / (1.f + expf(-sPart[tid]));
    sScore[tid] = s;
    scores[(size_t)b * T + t0 + tid] = s;
  }
  __syncthreads();

  // ---- denominator (Σ s_t) ----
  if (tid < 32) {
    float v = sScore[tid] + sScore[tid + 32];
    v += __shfl_xor(v, 16, 32);
    v += __shfl_xor(v, 8, 32);
    v += __shfl_xor(v, 4, 32);
    v += __shfl_xor(v, 2, 32);
    v += __shfl_xor(v, 1, 32);
    if (tid == 0) atomicAdd(&denom[b], v);
  }

  // ---- unnormalized context: ctx[b][e] += Σ_t s_t · value[t][e] (bf16 tile) ----
  {
    int e0 = tid * 4;
    float a0 = 0.f, a1 = 0.f, a2 = 0.f, a3 = 0.f;
    for (int t = 0; t < TM; ++t) {
      float s = sScore[t];
      uint2v v = *(const uint2v*)(sVal + t * ROWP + e0);
      a0 += s * bf2f(v.x & 0xffffu);
      a1 += s * bf2f(v.x >> 16);
      a2 += s * bf2f(v.y & 0xffffu);
      a3 += s * bf2f(v.y >> 16);
    }
    float* cp = ctx + (size_t)b * ENC + e0;
    atomicAdd(cp + 0, a0);
    atomicAdd(cp + 1, a1);
    atomicAdd(cp + 2, a2);
    atomicAdd(cp + 3, a3);
  }
}

// =====================================================================
// Finalize: context = ctx/denom ; align = scores/denom
// d_out = [context B*ENC][align B*T]
// =====================================================================
__global__ void finalize(const float* __restrict__ ctx, const float* __restrict__ scores,
                         const float* __restrict__ denom, float* __restrict__ out) {
  int i = blockIdx.x * blockDim.x + threadIdx.x;
  if (i < B * ENC) {
    out[i] = ctx[i] / denom[i / ENC];
  } else if (i < B * ENC + B * T) {
    int j = i - B * ENC;
    out[i] = scores[j] / denom[j / T];
  }
}

// =====================================================================
// Host entry
// =====================================================================
extern "C" void kernel_launch(void* const* d_in, const int* in_sizes, int n_in,
                              void* d_out, int out_size, void* d_ws, size_t ws_size,
                              hipStream_t stream) {
  (void)in_sizes; (void)n_in; (void)out_size; (void)ws_size;
  const float* query      = (const float*)d_in[0];
  const float* value      = (const float*)d_in[1];
  const float* last_align = (const float*)d_in[2];
  const float* conv_w     = (const float*)d_in[3];
  const float* conv_b     = (const float*)d_in[4];
  const float* WQ         = (const float*)d_in[5];
  const float* WV         = (const float*)d_in[6];
  const float* WU         = (const float*)d_in[7];
  const float* bias       = (const float*)d_in[8];
  const float* fc_w       = (const float*)d_in[9];
  float* out = (float*)d_out;

  // workspace layout (≈1.78 MB)
  char* ws = (char*)d_ws;
  unsigned short* WVb = (unsigned short*)ws;                 // 1,048,576 B
  float* scores = (float*)(ws + 1048576);                    //   524,288 B
  float* ctx    = (float*)(ws + 1572864);                    //   131,072 B
  float* denomv = (float*)(ws + 1703936);                    //       128 B
  float* qb2    = (float*)(ws + 1704064);                    //    65,536 B
  float* c0     = (float*)(ws + 1769600);                    //     2,048 B
  float* c1     = (float*)(ws + 1771648);                    //     2,048 B
  float* c2     = (float*)(ws + 1773696);                    //     2,048 B

  hipLaunchKernelGGL(prep_wv_bf16, dim3((ATTN * ENC / 4 + 255) / 256), dim3(256), 0, stream,
                     WV, WVb, ATTN * ENC / 4);
  hipLaunchKernelGGL(prep_qbc, dim3((B * ATTN + 255) / 256), dim3(256), 0, stream,
                     query, WQ, bias, WU, conv_w, conv_b, qb2, c0, c1, c2);
  hipLaunchKernelGGL(prep_zero, dim3((B * ENC + 255) / 256), dim3(256), 0, stream,
                     ctx, denomv);

  hipLaunchKernelGGL(law_main, dim3(T / TM, B), dim3(256), SMEM_MAIN, stream,
                     value, last_align, fc_w, WVb, qb2, c0, c1, c2,
                     scores, ctx, denomv);

  int total = B * ENC + B * T;
  hipLaunchKernelGGL(finalize, dim3((total + 255) / 256), dim3(256), 0, stream,
                     ctx, scores, denomv, out);
}
